// MaskedMMHCA_24043226923904
// MI455X (gfx1250) — matrix operand order repaired and checked by reading, not compiled, not run on hardware
//
#include <hip/hip_runtime.h>
#include <hip/hip_bf16.h>

// ---------------------------------------------------------------------------
// Problem constants (from reference): B=2, C=1024, T=2048, K=3, H=16, D=64
// ---------------------------------------------------------------------------
#define BB 2
#define CC 1024
#define TT 2048
#define HH 16
#define DD 64

typedef __attribute__((ext_vector_type(8)))  _Float16 v8h;
typedef __attribute__((ext_vector_type(16))) _Float16 v16h;
typedef __attribute__((ext_vector_type(8)))  float    v8f;

union V16 { v16h v; v8h h[2]; };

static __device__ __forceinline__ v8f wmma_f16(const V16& a, const V16& b, v8f c) {
    return __builtin_amdgcn_wmma_f32_16x16x32_f16(false, a.v, false, b.v,
                                                  (short)0, c, false, false);
}

// ---------------------------------------------------------------------------
// 1) fp32 -> fp16 weight conversion
// ---------------------------------------------------------------------------
__global__ void cvt_f16(const float* __restrict__ src, _Float16* __restrict__ dst, int n) {
    int i = blockIdx.x * 256 + threadIdx.x;
    if (i < n) dst[i] = (_Float16)src[i];
}

// ---------------------------------------------------------------------------
// 2) fused depthwise-conv(K=3) * mask + LayerNorm over C, for q/k/v branches.
//    One block per (b,t) column; 256 threads, 4 channels each.
//    Wave-level shfl reduction + one cross-wave LDS pass (1 barrier total).
// ---------------------------------------------------------------------------
__global__ __launch_bounds__(256) void dwln_kernel(
    const float* __restrict__ x, const unsigned char* __restrict__ mask,
    const float* __restrict__ wq, const float* __restrict__ wk, const float* __restrict__ wv,
    const float* __restrict__ qnw, const float* __restrict__ qnb,
    const float* __restrict__ knw, const float* __restrict__ knb,
    const float* __restrict__ vnw, const float* __restrict__ vnb,
    _Float16* __restrict__ qo, _Float16* __restrict__ ko, _Float16* __restrict__ vo)
{
    int bt  = blockIdx.x;
    int b   = bt / TT;
    int t   = bt % TT;
    int tid = threadIdx.x;
    int wid = tid >> 5, lane = tid & 31;
    float mf = mask[b * TT + t] ? 1.f : 0.f;

    float qv[4], kv[4], vv[4];
    float s[6] = {0.f, 0.f, 0.f, 0.f, 0.f, 0.f};

    #pragma unroll
    for (int i = 0; i < 4; ++i) {
        int c = tid + i * 256;
        const float* xr = x + ((size_t)b * CC + c) * TT;
        float xm = (t > 0)      ? xr[t - 1] : 0.f;
        float x0 = xr[t];
        float xp = (t < TT - 1) ? xr[t + 1] : 0.f;
        float q = (wq[c * 3 + 0] * xm + wq[c * 3 + 1] * x0 + wq[c * 3 + 2] * xp) * mf;
        float k = (wk[c * 3 + 0] * xm + wk[c * 3 + 1] * x0 + wk[c * 3 + 2] * xp) * mf;
        float v = (wv[c * 3 + 0] * xm + wv[c * 3 + 1] * x0 + wv[c * 3 + 2] * xp) * mf;
        qv[i] = q; kv[i] = k; vv[i] = v;
        s[0] += q; s[1] += q * q; s[2] += k; s[3] += k * k; s[4] += v; s[5] += v * v;
    }

    // intra-wave reduction (wave32)
    #pragma unroll
    for (int m = 16; m >= 1; m >>= 1)
        #pragma unroll
        for (int j = 0; j < 6; ++j) s[j] += __shfl_xor(s[j], m, 32);

    __shared__ float part[8][6];
    if (lane == 0)
        #pragma unroll
        for (int j = 0; j < 6; ++j) part[wid][j] = s[j];
    __syncthreads();
    float tot[6] = {0.f, 0.f, 0.f, 0.f, 0.f, 0.f};
    #pragma unroll
    for (int w = 0; w < 8; ++w)
        #pragma unroll
        for (int j = 0; j < 6; ++j) tot[j] += part[w][j];

    const float inv = 1.f / (float)CC;
    float qmu = tot[0] * inv, qrs = rsqrtf(tot[1] * inv - qmu * qmu + 1e-5f);
    float kmu = tot[2] * inv, krs = rsqrtf(tot[3] * inv - kmu * kmu + 1e-5f);
    float vmu = tot[4] * inv, vrs = rsqrtf(tot[5] * inv - vmu * vmu + 1e-5f);

    #pragma unroll
    for (int i = 0; i < 4; ++i) {
        int c = tid + i * 256;
        size_t o = ((size_t)b * CC + c) * TT + t;
        qo[o] = (_Float16)((qv[i] - qmu) * qrs * qnw[c] + qnb[c]);
        ko[o] = (_Float16)((kv[i] - kmu) * krs * knw[c] + knb[c]);
        vo[o] = (_Float16)((vv[i] - vmu) * vrs * vnw[c] + vnb[c]);
    }
}

// ---------------------------------------------------------------------------
// 3) Pointwise 1x1-conv GEMM: out[o,t] = sum_c W[o,c]*act[b,c,t] + bias[o]
//    Block tile 128(M=o) x 128(N=t); 8 waves; wave = 64x32 (4x2 WMMA tiles).
//    B tile double-buffered in LDS (one barrier per K step). A fragments use
//    loop-invariant integer offsets from the kernarg base so they lower to
//    global_load_b128 (NOT flat_load: flat couples LOADcnt with DScnt and
//    serializes against the LDS double-buffer stores).
// ---------------------------------------------------------------------------
__global__ __launch_bounds__(256) void gemm_pw(
    const _Float16* __restrict__ W,    // [C,C] row-major (o, c)
    const _Float16* __restrict__ act,  // [B][C][T]
    const float* __restrict__ bias,    // [C]
    _Float16* __restrict__ out16,      // [B][C][T] f16 output (or null)
    float* __restrict__ out32,         // [B][C][T] f32 output (or null)
    const unsigned char* __restrict__ mask, // [B][T] or null
    float alpha)
{
    int tb  = blockIdx.x * 128;
    int ob  = blockIdx.y * 128;
    int b   = blockIdx.z;
    int tid = threadIdx.x;
    int wid = tid >> 5;
    int lane = tid & 31;
    int ln = lane & 15, g = lane >> 4;
    int wm = (wid >> 2) * 64;   // wave M offset: 0 / 64
    int wn = (wid & 3) * 32;    // wave N offset: 0/32/64/96

    __shared__ _Float16 bt[2][128 * 32];   // [t][c], double-buffered, 16 KB
    __shared__ float bias_s[128];

    if (tid < 128) bias_s[tid] = bias[ob + tid];

    const _Float16* X = act + (size_t)b * CC * TT;

    v8f acc[4][2];
    #pragma unroll
    for (int mt = 0; mt < 4; ++mt)
        #pragma unroll
        for (int nt = 0; nt < 2; ++nt)
            #pragma unroll
            for (int e = 0; e < 8; ++e) acc[mt][nt][e] = 0.f;

    int cload = tid >> 3;          // 0..31: which k-row this thread stages
    int tseg  = (tid & 7) * 16;    // 16-element t segment
    const _Float16* bsrc = X + (size_t)cload * TT + tb + tseg;

    // loop-invariant A row offsets (element index into W); indexed by +kk
    size_t arow[4];
    #pragma unroll
    for (int mt = 0; mt < 4; ++mt)
        arow[mt] = (size_t)(ob + wm + mt * 16 + ln) * CC + 16 * g;

    // preload tile 0 into buffer 0
    {
        v8h u0 = *(const v8h*)(bsrc);
        v8h u1 = *(const v8h*)(bsrc + 8);
        #pragma unroll
        for (int j = 0; j < 8; ++j) {
            bt[0][(tseg + j) * 32 + cload]     = u0[j];
            bt[0][(tseg + 8 + j) * 32 + cload] = u1[j];
        }
    }
    int cur = 0;

    for (int kk = 0; kk < CC; kk += 32) {
        __syncthreads();

        bool hasNext = (kk + 32) < CC;
        v8h n0, n1;
        if (hasNext) {
            const _Float16* ns = bsrc + (size_t)(kk + 32) * TT;
            n0 = *(const v8h*)(ns);
            n1 = *(const v8h*)(ns + 8);
            if ((kk + 64) < CC)
                __builtin_prefetch(bsrc + (size_t)(kk + 64) * TT, 0, 1);
        }

        // A fragments for all 4 M tiles (global_load_b128 clause)
        V16 af[4];
        #pragma unroll
        for (int mt = 0; mt < 4; ++mt) {
            const _Float16* ap = W + (arow[mt] + kk);
            af[mt].h[0] = *(const v8h*)(ap);
            af[mt].h[1] = *(const v8h*)(ap + 8);
        }

        // B fragments: lane (g,n): K = 16g..16g+15 at column n
        const _Float16* bb = bt[cur];
        V16 bf[2];
        #pragma unroll
        for (int nt = 0; nt < 2; ++nt) {
            int col = wn + nt * 16 + ln;
            bf[nt].h[0] = *(const v8h*)&bb[col * 32 + 16 * g];
            bf[nt].h[1] = *(const v8h*)&bb[col * 32 + 16 * g + 8];
        }

        #pragma unroll
        for (int mt = 0; mt < 4; ++mt)
            #pragma unroll
            for (int nt = 0; nt < 2; ++nt)
                acc[mt][nt] = wmma_f16(af[mt], bf[nt], acc[mt][nt]);

        // store next tile into the other buffer (visible after next barrier)
        if (hasNext) {
            _Float16* nb = bt[cur ^ 1];
            #pragma unroll
            for (int j = 0; j < 8; ++j) {
                nb[(tseg + j) * 32 + cload]     = n0[j];
                nb[(tseg + 8 + j) * 32 + cload] = n1[j];
            }
        }
        cur ^= 1;
    }

    // epilogue: D layout lane(g,n): element r -> (m = r+8g, n)
    #pragma unroll
    for (int mt = 0; mt < 4; ++mt) {
        #pragma unroll
        for (int nt = 0; nt < 2; ++nt) {
            int t = tb + wn + nt * 16 + ln;
            float mf = 1.f;
            if (out32 && mask) mf = mask[b * TT + t] ? 1.f : 0.f;
            #pragma unroll
            for (int r = 0; r < 8; ++r) {
                int mo = wm + mt * 16 + r + 8 * g;
                float v = (acc[mt][nt][r] + bias_s[mo]) * alpha;
                size_t idx = ((size_t)b * CC + ob + mo) * TT + t;
                if (out32) out32[idx] = v * mf;
                else       out16[idx] = (_Float16)v;
            }
        }
    }
}

// ---------------------------------------------------------------------------
// 4) Flash attention per (b,h): 4 waves/block, 16 queries/wave, 32-key tiles,
//    online softmax. Scale folded into q projection. attn_out f16 [B][C][T].
//    K tile double-buffered (one barrier per step); V fragment loads issued
//    before the softmax so global latency overlaps the VALU section.
// ---------------------------------------------------------------------------
__global__ __launch_bounds__(128) void attn_kernel(
    const _Float16* __restrict__ qp, const _Float16* __restrict__ kp,
    const _Float16* __restrict__ vp, const unsigned char* __restrict__ mask,
    _Float16* __restrict__ attn_out)
{
    int qblk = blockIdx.x;          // 32 blocks of 64 queries
    int h    = blockIdx.y;
    int b    = blockIdx.z;
    int tid  = threadIdx.x;
    int wid  = tid >> 5;            // 0..3
    int lane = tid & 31;
    int ln = lane & 15, g = lane >> 4;

    __shared__ _Float16 q_lds[64 * 64];      // [tq][d]
    __shared__ _Float16 k_lds[2][32 * 64];   // [key][d], double-buffered
    __shared__ _Float16 p_lds[4][16 * 32];   // per-wave P scratch [m][k]

    const _Float16* qbase = qp + ((size_t)b * CC + h * DD) * TT;
    const _Float16* kbase = kp + ((size_t)b * CC + h * DD) * TT;
    const _Float16* vbase = vp + ((size_t)b * CC + h * DD) * TT;
    int q0 = qblk * 64;

    // stage Q [64 d][64 t] -> LDS [tq][d]
    for (int i = tid; i < 64 * 64 / 8; i += 128) {
        int d  = i >> 3;
        int t8 = (i & 7) * 8;
        v8h vv = *(const v8h*)(qbase + (size_t)d * TT + q0 + t8);
        #pragma unroll
        for (int j = 0; j < 8; ++j) q_lds[(t8 + j) * 64 + d] = vv[j];
    }

    // K staging geometry: this thread owns chunks (d, k8) and (d+32, k8)
    int sd = tid >> 2;              // 0..31 (d rows, +32 for second chunk)
    int k8 = (tid & 3) * 8;         // key sub-offset
    // preload K tile kk=0 into buffer 0
    {
        v8h a0 = *(const v8h*)(kbase + (size_t)sd * TT + k8);
        v8h a1 = *(const v8h*)(kbase + (size_t)(sd + 32) * TT + k8);
        #pragma unroll
        for (int j = 0; j < 8; ++j) {
            k_lds[0][(k8 + j) * 64 + sd]      = a0[j];
            k_lds[0][(k8 + j) * 64 + sd + 32] = a1[j];
        }
    }
    int cur = 0;
    __syncthreads();

    // Q A-fragments (K = d 0..31, 32..63), kept resident
    int qrow = wid * 16 + ln;
    V16 qa0, qa1;
    qa0.h[0] = *(const v8h*)&q_lds[qrow * 64 + 8 * g];
    qa0.h[1] = *(const v8h*)&q_lds[qrow * 64 + 16 + 8 * g];
    qa1.h[0] = *(const v8h*)&q_lds[qrow * 64 + 32 + 8 * g];
    qa1.h[1] = *(const v8h*)&q_lds[qrow * 64 + 48 + 8 * g];

    // loop-invariant V row offsets (indexed by +kk)
    size_t vrow[4];
    #pragma unroll
    for (int dt = 0; dt < 4; ++dt)
        vrow[dt] = (size_t)(dt * 16 + ln) * TT + 16 * g;

    float rm[8], rl[8];
    v8f o[4];
    #pragma unroll
    for (int r = 0; r < 8; ++r) { rm[r] = -3.0e38f; rl[r] = 0.f; }
    #pragma unroll
    for (int dt = 0; dt < 4; ++dt)
        #pragma unroll
        for (int e = 0; e < 8; ++e) o[dt][e] = 0.f;

    for (int kk = 0; kk < TT; kk += 32) {
        bool hasNext = (kk + 32) < TT;
        v8h a0, a1;
        if (hasNext) {
            a0 = *(const v8h*)(kbase + (size_t)sd * TT + kk + 32 + k8);
            a1 = *(const v8h*)(kbase + (size_t)(sd + 32) * TT + kk + 32 + k8);
        }

        // K B-fragments from current buffer: [key-half nh][d-half dh]
        const _Float16* kl = k_lds[cur];
        V16 kb[2][2];
        #pragma unroll
        for (int nh = 0; nh < 2; ++nh)
            #pragma unroll
            for (int dh = 0; dh < 2; ++dh) {
                int key = nh * 16 + ln;
                kb[nh][dh].h[0] = *(const v8h*)&kl[key * 64 + dh * 32 + 8 * g];
                kb[nh][dh].h[1] = *(const v8h*)&kl[key * 64 + dh * 32 + 16 + 8 * g];
            }

        // scores (scale already folded into q)
        v8f s0, s1;
        #pragma unroll
        for (int e = 0; e < 8; ++e) { s0[e] = 0.f; s1[e] = 0.f; }
        s0 = wmma_f16(qa0, kb[0][0], s0);
        s0 = wmma_f16(qa1, kb[0][1], s0);
        s1 = wmma_f16(qa0, kb[1][0], s1);
        s1 = wmma_f16(qa1, kb[1][1], s1);

        // V fragments: issue now so they are in flight during the softmax
        V16 vb[4];
        #pragma unroll
        for (int dt = 0; dt < 4; ++dt) {
            const _Float16* vptr = vbase + (vrow[dt] + kk);
            vb[dt].h[0] = *(const v8h*)(vptr);
            vb[dt].h[1] = *(const v8h*)(vptr + 8);
        }

        // key-padding mask (branchless selects)
        bool mk0 = mask[b * TT + kk + ln] != 0;
        bool mk1 = mask[b * TT + kk + 16 + ln] != 0;
        #pragma unroll
        for (int r = 0; r < 8; ++r) {
            s0[r] = mk0 ? s0[r] : -3.0e38f;
            s1[r] = mk1 ? s1[r] : -3.0e38f;
        }

        // online softmax: row reductions across the 16-lane half
        float tmax[8];
        #pragma unroll
        for (int r = 0; r < 8; ++r) tmax[r] = fmaxf(s0[r], s1[r]);
        #pragma unroll
        for (int m = 8; m >= 1; m >>= 1)
            #pragma unroll
            for (int r = 0; r < 8; ++r)
                tmax[r] = fmaxf(tmax[r], __shfl_xor(tmax[r], m, 16));

        float al[8], ps[8];
        #pragma unroll
        for (int r = 0; r < 8; ++r) {
            float mn = fmaxf(rm[r], tmax[r]);
            al[r] = __expf(rm[r] - mn);
            rm[r] = mn;
            s0[r] = __expf(s0[r] - mn);
            s1[r] = __expf(s1[r] - mn);
            ps[r] = s0[r] + s1[r];
        }
        #pragma unroll
        for (int m = 8; m >= 1; m >>= 1)
            #pragma unroll
            for (int r = 0; r < 8; ++r) ps[r] += __shfl_xor(ps[r], m, 16);
        #pragma unroll
        for (int r = 0; r < 8; ++r) rl[r] = rl[r] * al[r] + ps[r];
        #pragma unroll
        for (int dt = 0; dt < 4; ++dt)
            #pragma unroll
            for (int r = 0; r < 8; ++r) o[dt][r] *= al[r];

        // P (D-layout) -> A-fragment layout via per-wave LDS
        _Float16* pl = p_lds[wid];
        #pragma unroll
        for (int r = 0; r < 8; ++r) {
            pl[(r + 8 * g) * 32 + ln]      = (_Float16)s0[r];
            pl[(r + 8 * g) * 32 + 16 + ln] = (_Float16)s1[r];
        }
        asm volatile("s_wait_dscnt 0" ::: "memory");
        V16 pa;
        pa.h[0] = *(const v8h*)&pl[ln * 32 + 16 * g];
        pa.h[1] = *(const v8h*)&pl[ln * 32 + 16 * g + 8];

        #pragma unroll
        for (int dt = 0; dt < 4; ++dt)
            o[dt] = wmma_f16(pa, vb[dt], o[dt]);

        // publish next K tile to the other buffer; barrier flips buffers
        if (hasNext) {
            _Float16* nb = k_lds[cur ^ 1];
            #pragma unroll
            for (int j = 0; j < 8; ++j) {
                nb[(k8 + j) * 64 + sd]      = a0[j];
                nb[(k8 + j) * 64 + sd + 32] = a1[j];
            }
            __syncthreads();
        }
        cur ^= 1;
    }

    // normalize + store f16 [b][h*64+d][t]
    #pragma unroll
    for (int dt = 0; dt < 4; ++dt) {
        size_t cbase = (size_t)b * CC + h * DD + dt * 16 + ln;
        #pragma unroll
        for (int r = 0; r < 8; ++r) {
            float val = o[dt][r] / rl[r];
            attn_out[cbase * TT + q0 + wid * 16 + r + 8 * g] = (_Float16)val;
        }
    }
}

// second tuple output: mask passthrough (written as 0/1 in d_out's tail)
__global__ void mask_tail(const unsigned char* __restrict__ mask, float* __restrict__ outf, int n) {
    int i = blockIdx.x * 256 + threadIdx.x;
    if (i < n) outf[i] = mask[i] ? 1.f : 0.f;
}

// ---------------------------------------------------------------------------
extern "C" void kernel_launch(void* const* d_in, const int* in_sizes, int n_in,
                              void* d_out, int out_size, void* d_ws, size_t ws_size,
                              hipStream_t stream) {
    const float*         x     = (const float*)d_in[0];
    const unsigned char* mask  = (const unsigned char*)d_in[1];
    const float* wq_dw = (const float*)d_in[2];
    const float* wk_dw = (const float*)d_in[3];
    const float* wv_dw = (const float*)d_in[4];
    const float* qn_w = (const float*)d_in[5],  *qn_b = (const float*)d_in[6];
    const float* kn_w = (const float*)d_in[7],  *kn_b = (const float*)d_in[8];
    const float* vn_w = (const float*)d_in[9],  *vn_b = (const float*)d_in[10];
    const float* w_q = (const float*)d_in[11],  *b_q = (const float*)d_in[12];
    const float* w_k = (const float*)d_in[13],  *b_k = (const float*)d_in[14];
    const float* w_v = (const float*)d_in[15],  *b_v = (const float*)d_in[16];
    const float* w_p = (const float*)d_in[17],  *b_p = (const float*)d_in[18];

    const size_t CCn = (size_t)CC * CC;          // 1M elems
    const size_t ACT = (size_t)BB * CC * TT;     // 4M elems

    char* ws = (char*)d_ws;
    _Float16* wq16 = (_Float16*)(ws + 0 * CCn * 2);
    _Float16* wk16 = (_Float16*)(ws + 1 * CCn * 2);
    _Float16* wv16 = (_Float16*)(ws + 2 * CCn * 2);
    _Float16* wp16 = (_Float16*)(ws + 3 * CCn * 2);
    char* p = ws + 4 * CCn * 2;
    _Float16* qln = (_Float16*)(p + 0 * ACT * 2);
    _Float16* kln = (_Float16*)(p + 1 * ACT * 2);
    _Float16* vln = (_Float16*)(p + 2 * ACT * 2);
    _Float16* qpb = (_Float16*)(p + 3 * ACT * 2);
    _Float16* kpb = (_Float16*)(p + 4 * ACT * 2);
    _Float16* vpb = (_Float16*)(p + 5 * ACT * 2);
    _Float16* att = (_Float16*)(p + 6 * ACT * 2);

    // 1) weights -> f16
    cvt_f16<<<(int)(CCn / 256), 256, 0, stream>>>(w_q, wq16, (int)CCn);
    cvt_f16<<<(int)(CCn / 256), 256, 0, stream>>>(w_k, wk16, (int)CCn);
    cvt_f16<<<(int)(CCn / 256), 256, 0, stream>>>(w_v, wv16, (int)CCn);
    cvt_f16<<<(int)(CCn / 256), 256, 0, stream>>>(w_p, wp16, (int)CCn);

    // 2) fused depthwise conv + mask + layernorm
    dwln_kernel<<<BB * TT, 256, 0, stream>>>(x, mask, wq_dw, wk_dw, wv_dw,
                                             qn_w, qn_b, kn_w, kn_b, vn_w, vn_b,
                                             qln, kln, vln);

    // 3) q/k/v projections (scale 1/sqrt(D) folded into q)
    dim3 gg(TT / 128, CC / 128, BB);
    gemm_pw<<<gg, 256, 0, stream>>>(wq16, qln, b_q, qpb, nullptr, nullptr, 0.125f);
    gemm_pw<<<gg, 256, 0, stream>>>(wk16, kln, b_k, kpb, nullptr, nullptr, 1.0f);
    gemm_pw<<<gg, 256, 0, stream>>>(wv16, vln, b_v, vpb, nullptr, nullptr, 1.0f);

    // 4) flash attention
    attn_kernel<<<dim3(TT / 64, HH, BB), 128, 0, stream>>>(qpb, kpb, vpb, mask, att);

    // 5) output projection, f32 + mask
    gemm_pw<<<gg, 256, 0, stream>>>(wp16, att, b_p, nullptr, (float*)d_out, mask, 1.0f);

    // 6) tuple second output: mask passthrough
    int tail = out_size - (int)ACT;
    if (tail > 0) {
        float* outf = (float*)d_out + ACT;
        mask_tail<<<(tail + 255) / 256, 256, 0, stream>>>(mask, outf, tail);
    }
}
